// SpikingNet_23871428231449
// MI455X (gfx1250) — matrix operand order repaired and checked by reading, not compiled, not run on hardware
//
#include <hip/hip_runtime.h>

// ---------------------------------------------------------------------------
// Spiking net, persistent-timeloop WMMA kernel for gfx1250 (MI455X).
//   grid = 256 blocks (16 batch rows each), block = 256 threads (8 wave32).
//   waves 0..6 : layer1 GEMM tile (16 x 16 of H1=100 padded to 112), LIF state
//   wave  7    : layer2 GEMM (K=100 padded 128, N=10 padded 16), LIF + sum
// The per-block image tile is brought into LDS with the Tensor Data Mover
// (tensor_load_to_lds + s_wait_tensorcnt); weights are converted f32->f16 by
// VALU into LDS. The 100-step recurrence then runs entirely on-chip.
// ---------------------------------------------------------------------------

typedef __attribute__((ext_vector_type(16))) _Float16 v16h;
typedef __attribute__((ext_vector_type(8)))  _Float16 v8h;
typedef __attribute__((ext_vector_type(8)))  float    v8f;
typedef __attribute__((ext_vector_type(4)))  unsigned u32x4;
typedef __attribute__((ext_vector_type(4)))  int      i32x4;
typedef __attribute__((ext_vector_type(8)))  int      i32x8;

#define N_IN   784
#define H1     100
#define H2     10
#define T_STEPS 100
#define BEGIN  20
#define BATCH  4096
#define DECAY  0.9f

#define K1P    800     // N_IN padded to multiple of 32
#define N1P    112     // H1 padded to 7*16
#define K2P    128     // H1-as-K padded to multiple of 32
#define MT     16      // batch rows per workgroup

// LDS byte offsets (all 16B aligned). lds_x MUST stay at offset 0: with no
// static __shared__ data, dynamic LDS starts at LDS byte 0, and the TDM
// descriptor needs the raw LDS address of the tile destination.
#define OFF_X     0                          // float  [16][784]   = 50176
#define OFF_SUM   (OFF_X   + MT*N_IN*4)      // float  [16][16]    = 1024
#define OFF_W1T   (OFF_SUM + 16*16*4)        // f16    [112][800]  = 179200
#define OFF_W2T   (OFF_W1T + N1P*K1P*2)      // f16    [16][128]   = 4096
#define OFF_XI    (OFF_W2T + 16*K2P*2)       // f16    [16][800]   = 25600
#define OFF_OUT1  (OFF_XI  + MT*K1P*2)       // f16    [16][128]   = 4096
#define SMEM_BYTES (OFF_OUT1 + MT*K2P*2)     // total = 264192 (< 320KB/WGP)

__device__ __forceinline__ unsigned mix32(unsigned s) {
    s ^= s >> 16; s *= 0x7feb352du;
    s ^= s >> 15; s *= 0x846ca68bu;
    s ^= s >> 16;
    return s;
}

// Issue a TDM 2D tile load: x tile [MT rows][N_IN floats] -> LDS offset OFF_X.
// D# built per CDNA5 ISA ch.8: group0 = {count, lds_addr, global_addr, type},
// group1 = {data_size, tensor dims, tile dims, dim0 stride}.
__device__ __forceinline__ void tdm_load_x_tile(const float* gsrc) {
    unsigned long long ga = (unsigned long long)(const void*)gsrc;
    u32x4 g0;
    g0[0] = 1u;                                          // count=1, user desc
    g0[1] = (unsigned)OFF_X;                             // lds_addr (bytes)
    g0[2] = (unsigned)(ga & 0xFFFFFFFFu);                // global_addr[31:0]
    g0[3] = (unsigned)((ga >> 32) & 0x1FFFFFFu)          // global_addr[56:32]
          | (2u << 30);                                  // type = 2 ("image")
    i32x8 g1;
    g1[0] = (int)(2u << 16);        // wg_mask=0, data_size=2 (4 bytes)
    g1[1] = (int)((unsigned)N_IN << 16);   // tensor_dim0[15:0]  (bits 63:48)
    g1[2] = (int)((unsigned)MT   << 16);   // dim0 hi=0 | tensor_dim1 lo = 16
    g1[3] = (int)((unsigned)N_IN << 16);   // dim1 hi=0 | tile_dim0 = 784
    g1[4] = MT;                     // tile_dim1 = 16, tile_dim2 = 0
    g1[5] = N_IN;                   // tensor_dim0_stride[31:0] = 784
    g1[6] = 0;
    g1[7] = 0;
    i32x4 gz4 = {0, 0, 0, 0};
#if __has_include(<hip/amd_detail/amd_gfx1250_TDM.h>)
    // therock-10.0 headers -> 6-arg builtin
    i32x8 gz8 = {0, 0, 0, 0, 0, 0, 0, 0};
    __builtin_amdgcn_tensor_load_to_lds(g0, g1, gz4, gz4, gz8, 0);
#else
    // ROCm 7.2 (clang-22) -> 5-arg builtin
    __builtin_amdgcn_tensor_load_to_lds(g0, g1, gz4, gz4, 0);
#endif
}

__global__ __launch_bounds__(256) void
snn_persistent_wmma(const float* __restrict__ x,
                    const float* __restrict__ W1,
                    const float* __restrict__ b1,
                    const float* __restrict__ W2,
                    const float* __restrict__ b2,
                    float* __restrict__ out) {
    extern __shared__ char smem[];
    float*    lds_x    = (float*)   (smem + OFF_X);
    float*    lds_sum  = (float*)   (smem + OFF_SUM);
    _Float16* lds_w1t  = (_Float16*)(smem + OFF_W1T);   // [n (112)][k (800)]
    _Float16* lds_w2t  = (_Float16*)(smem + OFF_W2T);   // [n (16)][k (128)]
    _Float16* lds_xi   = (_Float16*)(smem + OFF_XI);    // [m (16)][k (800)]
    _Float16* lds_out1 = (_Float16*)(smem + OFF_OUT1);  // [m (16)][n (128)]

    const int tid  = threadIdx.x;
    const int wave = tid >> 5;
    const int lane = tid & 31;
    const int ln   = lane & 15;     // N (or M for A-frag) index within tile
    const int h    = lane >> 4;     // half-wave select
    const int m0   = blockIdx.x * MT;

    // Async image-tile fetch via the Tensor Data Mover (wave 0 issues; TDM
    // ignores EXEC, one descriptor per wave is enough).
    if (wave == 0)
        tdm_load_x_tile(x + (size_t)m0 * N_IN);

    // L2 warm hint for the weight matrix (global_prefetch_b8 path)
    __builtin_prefetch(W1 + tid * 256, 0, 0);

    // ---- one-time LDS population (weights f32->f16, zero pads) -------------
    #pragma unroll 1
    for (int i = tid; i < N1P * K1P; i += 256) {     // W1^T, f16, zero padded
        int n = i / K1P, k = i - n * K1P;
        lds_w1t[i] = (n < H1 && k < N_IN) ? (_Float16)W1[k * H1 + n] : (_Float16)0.f;
    }
    #pragma unroll 1
    for (int i = tid; i < 16 * K2P; i += 256) {      // W2^T, f16, zero padded
        int n = i / K2P, k = i - n * K2P;
        lds_w2t[i] = (n < H2 && k < H1) ? (_Float16)W2[k * H2 + n] : (_Float16)0.f;
    }
    #pragma unroll 1
    for (int i = tid; i < MT * K2P; i += 256)        // delayed spikes start at 0
        lds_out1[i] = (_Float16)0.f;
    #pragma unroll 1
    for (int i = tid; i < MT * (K1P - N_IN); i += 256) { // xi K-pad stays 0
        int m = i / (K1P - N_IN), k = N_IN + (i - m * (K1P - N_IN));
        lds_xi[m * K1P + k] = (_Float16)0.f;
    }
    if (wave == 0)
        __builtin_amdgcn_s_wait_tensorcnt(0);        // x tile landed in LDS
    __syncthreads();

    // ---- persistent per-lane state -----------------------------------------
    float inner1[8] = {0.f,0.f,0.f,0.f,0.f,0.f,0.f,0.f};
    float inner2[8] = {0.f,0.f,0.f,0.f,0.f,0.f,0.f,0.f};
    float sum2[8]   = {0.f,0.f,0.f,0.f,0.f,0.f,0.f,0.f};
    float b1n = 0.f, b2n = 0.f;
    if (wave < 7) { int n = wave * 16 + ln; b1n = (n < H1) ? b1[n] : 0.f; }
    else          { b2n = (ln < H2) ? b2[ln] : 0.f; }

    // ---- time recurrence ----------------------------------------------------
    #pragma unroll 1
    for (int t = 0; t < T_STEPS; ++t) {
        // (a) random input gating -> xi (f16), cooperatively, 2 elems/iter
        #pragma unroll 1
        for (int e = tid; e < MT * (N_IN / 2); e += 256) {
            int m = e / (N_IN / 2);
            int k = (e - m * (N_IN / 2)) * 2;
            unsigned s = (unsigned)((m0 + m) * N_IN + k) * 2654435761u
                       + (unsigned)t * 0x9E3779B9u;
            s = mix32(s);
            float r0 = (float)(s & 0xFFFFu) * (1.f / 65536.f);
            float r1 = (float)(s >> 16)     * (1.f / 65536.f);
            union { _Float16 hv[2]; unsigned u; } p;
            p.hv[0] = (_Float16)(r0 * lds_x[m * N_IN + k]);
            p.hv[1] = (_Float16)(r1 * lds_x[m * N_IN + k + 1]);
            *(unsigned*)&lds_xi[m * K1P + k] = p.u;     // k even -> 4B aligned
        }
        __syncthreads();   // xi ready; prev-step lds_out1 stable

        float outs[8];
        if (wave < 7) {
            // ---- layer 1: [16 x 800] x [800 x 16] on this wave's N tile ----
            v8f c;
            #pragma unroll
            for (int r = 0; r < 8; ++r) c[r] = inner1[r] * DECAY + b1n;

            const _Float16* arow = lds_xi  + ln * K1P + 8 * h;
            const _Float16* brow = lds_w1t + (wave * 16 + ln) * K1P + 8 * h;
            #pragma unroll
            for (int kk = 0; kk < K1P / 32; ++kk) {
                v16h a, b;
                ((v8h*)&a)[0] = *(const v8h*)(arow + 32 * kk);       // K = 32kk+8h ..
                ((v8h*)&a)[1] = *(const v8h*)(arow + 32 * kk + 16);  // K = +16 ..
                ((v8h*)&b)[0] = *(const v8h*)(brow + 32 * kk);       // K = 32kk+8h ..
                ((v8h*)&b)[1] = *(const v8h*)(brow + 32 * kk + 16);
                c = __builtin_amdgcn_wmma_f32_16x16x32_f16(
                        false, a, false, b, (short)0, c, false, false);
            }
            // ---- LIF update: relu spike, penalty reset ----
            #pragma unroll
            for (int r = 0; r < 8; ++r) {
                float in = c[r];
                float ov = fmaxf(in - 1.0f, 0.f);          // THRESH = 1
                inner1[r] = (ov > 0.f) ? -0.5f * in : in;  // 1 - PEN/THRESH
                outs[r] = ov;
            }
        } else {
            // ---- layer 2 on delayed spikes: [16 x 128] x [128 x 16] ----
            v8f c;
            #pragma unroll
            for (int r = 0; r < 8; ++r) c[r] = inner2[r] * DECAY + b2n;

            const _Float16* arow = lds_out1 + ln * K2P + 8 * h;
            const _Float16* brow = lds_w2t  + ln * K2P + 8 * h;
            #pragma unroll
            for (int kk = 0; kk < K2P / 32; ++kk) {
                v16h a, b;
                ((v8h*)&a)[0] = *(const v8h*)(arow + 32 * kk);
                ((v8h*)&a)[1] = *(const v8h*)(arow + 32 * kk + 16);
                ((v8h*)&b)[0] = *(const v8h*)(brow + 32 * kk);
                ((v8h*)&b)[1] = *(const v8h*)(brow + 32 * kk + 16);
                c = __builtin_amdgcn_wmma_f32_16x16x32_f16(
                        false, a, false, b, (short)0, c, false, false);
            }
            #pragma unroll
            for (int r = 0; r < 8; ++r) {
                float in = c[r];
                bool fire = (in - 1.0f) > 0.f;
                in = fire ? -0.5f * in : in;
                inner2[r] = in;
                if (t >= BEGIN - 1 && t <= T_STEPS - 2) sum2[r] += in;
            }
        }
        __syncthreads();   // wave7 done reading lds_out1; publish new spikes

        if (wave < 7) {
            #pragma unroll
            for (int r = 0; r < 8; ++r)
                lds_out1[(r + 8 * h) * K2P + wave * 16 + ln] = (_Float16)outs[r];
        }
    }

    // ---- log_softmax over the 10 classes -----------------------------------
    if (wave == 7) {
        #pragma unroll
        for (int r = 0; r < 8; ++r)
            lds_sum[(r + 8 * h) * 16 + ln] = sum2[r];
    }
    __syncthreads();
    if (tid < MT) {
        float v[H2], mx = -3.4e38f;
        #pragma unroll
        for (int n = 0; n < H2; ++n) { v[n] = lds_sum[tid * 16 + n]; mx = fmaxf(mx, v[n]); }
        float se = 0.f;
        #pragma unroll
        for (int n = 0; n < H2; ++n) se += expf(v[n] - mx);
        float lse = logf(se);
        #pragma unroll
        for (int n = 0; n < H2; ++n)
            out[(m0 + tid) * H2 + n] = v[n] - mx - lse;
    }
}

extern "C" void kernel_launch(void* const* d_in, const int* in_sizes, int n_in,
                              void* d_out, int out_size, void* d_ws, size_t ws_size,
                              hipStream_t stream) {
    (void)in_sizes; (void)n_in; (void)d_ws; (void)ws_size; (void)out_size;
    const float* x  = (const float*)d_in[0];
    const float* W1 = (const float*)d_in[1];
    const float* b1 = (const float*)d_in[2];
    const float* W2 = (const float*)d_in[3];
    const float* b2 = (const float*)d_in[4];
    float* out = (float*)d_out;

    dim3 grid(BATCH / MT);   // 256 persistent workgroups
    dim3 block(256);         // 8 wave32 per workgroup
    snn_persistent_wmma<<<grid, block, SMEM_BYTES, stream>>>(x, W1, b1, W2, b2, out);
}
